// MonarchProj_14542759265068
// MI455X (gfx1250) — matrix operand order repaired and checked by reading, not compiled, not run on hardware
//
#include <hip/hip_runtime.h>
#include <hip/hip_bf16.h>

#define IN_DIM 4096
#define BS 64
#define NB 64            // blocks per dim (4096/64)
#define M_TILE 16        // tokens per workgroup
#define ROWP 4104        // padded y row stride in bf16 elems (8208 B, 16B-aligned)

// LDS layout: [ xbuf ping | xbuf pong | ylds ]
#define XROW_DW    68                      // 64 data + 4 pad DWORDs per x row (TDM pad feature)
#define XCHUNK_DW  (8 * 16 * XROW_DW)      // 8 blocks x 16 rows -> 8704 DW = 34816 B
#define XBUF_BYTES (XCHUNK_DW * 4)
#define YLDS_OFF   (2 * XBUF_BYTES)        // 69632 B
#define LDS_BYTES  (YLDS_OFF + M_TILE * ROWP * 2)   // 200960 B (< 320 KB WGP LDS)

#define WS_BYTES   (2u * 64u * 4u * 2u * 32u * 8u * 4u)   // 1 MB of preconverted B fragments

typedef __attribute__((ext_vector_type(16))) __bf16        v16bf;
typedef __attribute__((ext_vector_type(8)))  float         v8f;
typedef __attribute__((ext_vector_type(8)))  unsigned int  v8u;
typedef __attribute__((ext_vector_type(4)))  unsigned int  u32x4;
typedef __attribute__((ext_vector_type(4)))  int           i32x4;
typedef __attribute__((ext_vector_type(8)))  int           i32x8;

#if defined(__has_builtin)
#if __has_builtin(__builtin_amdgcn_tensor_load_to_lds)
#define HAVE_TDM 1
#endif
#endif
#ifndef HAVE_TDM
#define HAVE_TDM 0
#endif

// pack two fp32 -> bf16x2 (round-half-up), lo in [15:0]: 2x v_add + 1x v_perm_b32
static __device__ __forceinline__ unsigned pk_bf16(float lo, float hi) {
    union { float f; unsigned u; } a, b;
    a.f = lo; b.f = hi;
    return __builtin_amdgcn_perm(b.u + 0x8000u, a.u + 0x8000u, 0x07060302u);
}

static __device__ __forceinline__ unsigned pk_bf16_u(unsigned lo, unsigned hi) {
    return __builtin_amdgcn_perm(hi + 0x8000u, lo + 0x8000u, 0x07060302u);
}

static __device__ __forceinline__ unsigned short f32_to_bf16(float f) {
    union { float f; unsigned u; } c; c.f = f;
    return (unsigned short)((c.u + 0x8000u) >> 16);   // folds into ds_store_b16_d16_hi
}

#if HAVE_TDM
// TDM: load one x chunk = 3D tile {X=64 floats, Y=16 rows (stride 4096), Z=8 blocks (stride 64)}
// into LDS with 4-DW padding after every 64 DW -> LDS row stride 68 DW (4-bank stagger).
static __device__ __forceinline__ void tdm_load_x_chunk(const float* gsrc, unsigned lds_off) {
    unsigned long long ga = (unsigned long long)(const void*)gsrc;
    u32x4 g0;
    g0[0] = 1u;                                   // count=1 (user descriptor)
    g0[1] = lds_off;                              // lds_addr (bytes)
    g0[2] = (unsigned)ga;                         // global_addr[31:0]
    g0[3] = (unsigned)(ga >> 32) | (2u << 30);    // global_addr[56:32] | type=2 ("image")
    i32x8 g1;
    g1[0] = (int)((2u << 16) | (1u << 20) | (5u << 22) | (3u << 25));
           // data_size=4B | pad_enable | pad_interval: 64 DW | pad_amount: 4 DW
    g1[1] = (int)(64u << 16);                     // tensor_dim0 = 64 (low 16)
    g1[2] = (int)(16u << 16);                     // tensor_dim0 hi=0 | tensor_dim1 = 16 (low 16)
    g1[3] = (int)(64u << 16);                     // tensor_dim1 hi=0 | tile_dim0 = 64
    g1[4] = (int)(16u | (8u << 16));              // tile_dim1 = 16 | tile_dim2 = 8
    g1[5] = (int)IN_DIM;                          // tensor_dim0_stride = 4096 (low 32)
    g1[6] = (int)(64u << 16);                     // stride0 hi=0 | tensor_dim1_stride = 64 (low 16)
    g1[7] = 0;
    i32x4 g2;
    g2[0] = 8;                                    // tensor_dim2 = 8
    g2[1] = 0; g2[2] = 0; g2[3] = 0;
    i32x4 g3; g3[0] = 0; g3[1] = 0; g3[2] = 0; g3[3] = 0;
#if defined(__clang_major__) && (__clang_major__ >= 23)
    i32x8 g4; g4[0]=0; g4[1]=0; g4[2]=0; g4[3]=0; g4[4]=0; g4[5]=0; g4[6]=0; g4[7]=0;
    __builtin_amdgcn_tensor_load_to_lds(g0, g1, g2, g3, g4, 0);
#else
    __builtin_amdgcn_tensor_load_to_lds(g0, g1, g2, g3, 0);
#endif
}
#endif

// ---------------------------------------------------------------------------
// Pre-pass: convert R and L (fp32, row-major 64x64 per block) into bf16 packed
// in the exact WMMA B-fragment register layout.
// fragment id f = ((mat*64 + blk)*4 + nn)*2 + kkidx ; per fragment: 32 lanes x 8 dwords.
// ---------------------------------------------------------------------------
__global__ __launch_bounds__(256, 1)
void preconv_b_kernel(const float* __restrict__ R, const float* __restrict__ L,
                      unsigned* __restrict__ ws)
{
    const int gid   = blockIdx.x;        // 0..127
    const int mat   = gid >> 6;          // 0 = R, 1 = L
    const int i     = gid & 63;
    const float* src = (mat ? L : R) + (long)i * (BS * BS);

    const int tid   = threadIdx.x;
    const int lane  = tid & 31;
    const int wave  = tid >> 5;          // 8 waves -> 8 (nn, kkidx) combos
    const int nn    = wave >> 1;
    const int kkidx = wave & 1;
    const int n     = nn * 16 + (lane & 15);
    const int kbase = kkidx * 32 + (lane >> 4) * 16;

    v8u u;
    #pragma unroll
    for (int v = 0; v < 8; ++v) {
        float f0 = src[(kbase + 2 * v    ) * BS + n];
        float f1 = src[(kbase + 2 * v + 1) * BS + n];
        u[v] = pk_bf16(f0, f1);
    }
    const int f = ((mat * 64 + i) * 4 + nn) * 2 + kkidx;
    ((v8u*)ws)[f * 32 + lane] = u;
}

// ---------------------------------------------------------------------------
// Fused Monarch kernel.
// ---------------------------------------------------------------------------
template <bool USE_WS>
__global__ __launch_bounds__(256, 1)
void monarch_fused_kernel(const float* __restrict__ x,
                          const float* __restrict__ R,
                          const float* __restrict__ L,
                          const float* __restrict__ bias,
                          const unsigned* __restrict__ wsfrag,
                          float* __restrict__ out)
{
    extern __shared__ char smem[];
    float*          xbuf = (float*)smem;                            // 2 x TDM chunk
    unsigned short* ylds = (unsigned short*)(smem + YLDS_OFF);      // permuted y, bf16

    const int tid  = threadIdx.x;
    const int lane = tid & 31;
    const int wave = tid >> 5;          // 0..7
    const int mrow = lane & 15;
    const int sel  = lane >> 4;
    const long tok0 = (long)blockIdx.x * M_TILE;

    const float* xrow = x + (tok0 + mrow) * (long)IN_DIM;
    const v8u*   wsv  = (const v8u*)wsfrag;

    // ================= Stage 1: y(m,i,e) = x(m, i*64+:) @ R[i], scattered as
    // ylds[m][e*64 + i]  (the Monarch stride permutation applied on the fly)
#if HAVE_TDM
    if (wave == 0)
        tdm_load_x_chunk(x + tok0 * (long)IN_DIM, 0u);
#endif
    for (int ib = 0; ib < NB / 8; ++ib) {
        const int i = wave + ib * 8;                  // this wave's input block

#if HAVE_TDM
        if (ib + 1 < NB / 8) {
            if (wave == 0) {
                tdm_load_x_chunk(x + tok0 * (long)IN_DIM + (long)(ib + 1) * 512,
                                 (unsigned)(((ib + 1) & 1) * XBUF_BYTES));
                __builtin_amdgcn_s_wait_tensorcnt(1);   // oldest (current chunk) done
            }
        } else {
            if (wave == 0) __builtin_amdgcn_s_wait_tensorcnt(0);
        }
        __syncthreads();                               // publish chunk ib to all waves
        const float* xc = xbuf + (size_t)((ib & 1) * XCHUNK_DW);
#else
        if (ib + 1 < NB / 8)
            __builtin_prefetch(xrow + (size_t)(i + 8) * BS, 0, 0);
#endif
        const float* Rb = R + (long)i * (BS * BS);

        v8f acc[4];
        #pragma unroll
        for (int nn = 0; nn < 4; ++nn) {
            v8f z = {0.f,0.f,0.f,0.f,0.f,0.f,0.f,0.f};
            acc[nn] = z;
        }

        #pragma unroll
        for (int kk = 0; kk < BS; kk += 32) {
            // ---- A (16x32 bf16) built directly in WMMA VGPR layout
#if HAVE_TDM
            const float* ap = xc + (wave * 16 + mrow) * XROW_DW + kk + sel * 16;
#else
            const float* ap = xrow + i * BS + kk + sel * 16;
#endif
            uint4 q0 = *(const uint4*)(ap + 0);
            uint4 q1 = *(const uint4*)(ap + 4);
            uint4 q2 = *(const uint4*)(ap + 8);
            uint4 q3 = *(const uint4*)(ap + 12);
            v8u au;
            au[0] = pk_bf16_u(q0.x, q0.y); au[1] = pk_bf16_u(q0.z, q0.w);
            au[2] = pk_bf16_u(q1.x, q1.y); au[3] = pk_bf16_u(q1.z, q1.w);
            au[4] = pk_bf16_u(q2.x, q2.y); au[5] = pk_bf16_u(q2.z, q2.w);
            au[6] = pk_bf16_u(q3.x, q3.y); au[7] = pk_bf16_u(q3.z, q3.w);
            v16bf a = __builtin_bit_cast(v16bf, au);

            const int kkidx = kk >> 5;
            const int kbase = kk + sel * 16;
            #pragma unroll
            for (int nn = 0; nn < 4; ++nn) {
                v8u bu;
                if constexpr (USE_WS) {
                    const int f = (i * 4 + nn) * 2 + kkidx;         // mat = 0 (R)
                    bu = wsv[f * 32 + lane];
                } else {
                    const int n = nn * 16 + mrow;
                    #pragma unroll
                    for (int v = 0; v < 8; ++v) {
                        float f0 = Rb[(kbase + 2 * v    ) * BS + n];
                        float f1 = Rb[(kbase + 2 * v + 1) * BS + n];
                        bu[v] = pk_bf16(f0, f1);
                    }
                }
                v16bf b = __builtin_bit_cast(v16bf, bu);
                acc[nn] = __builtin_amdgcn_wmma_f32_16x16x32_bf16(
                    false, a, false, b, (short)0, acc[nn], false, false);
            }
        }

        // scatter y(m, i, e) -> ylds[m][e*64 + i]
        #pragma unroll
        for (int nn = 0; nn < 4; ++nn) {
            const int e = nn * 16 + mrow;
            #pragma unroll
            for (int r = 0; r < 8; ++r) {
                const int m = r + sel * 8;
                ylds[m * ROWP + e * BS + i] = f32_to_bf16(acc[nn][r]);
            }
        }
#if HAVE_TDM
        __syncthreads();   // all waves done reading xbuf[ib&1] before TDM reuses it
#endif
    }

    __syncthreads();

    // ================= Stage 2: out(m, o*64+c) = y_perm(m,o,:) @ L[o] + bias
    for (int ob = 0; ob < NB / 8; ++ob) {
        const int o = wave + ob * 8;
        const float* Lb = L + (long)o * (BS * BS);

        v8f acc[4];
        #pragma unroll
        for (int nn = 0; nn < 4; ++nn) {
            v8f z = {0.f,0.f,0.f,0.f,0.f,0.f,0.f,0.f};
            acc[nn] = z;
        }

        #pragma unroll
        for (int kk = 0; kk < BS; kk += 32) {
            // ---- A from LDS, already bf16-packed + permuted: two ds_load_b128
            const unsigned short* ap = ylds + mrow * ROWP + o * BS + kk + sel * 16;
            uint4 d0 = *(const uint4*)(ap + 0);
            uint4 d1 = *(const uint4*)(ap + 8);
            v8u au;
            au[0] = d0.x; au[1] = d0.y; au[2] = d0.z; au[3] = d0.w;
            au[4] = d1.x; au[5] = d1.y; au[6] = d1.z; au[7] = d1.w;
            v16bf a = __builtin_bit_cast(v16bf, au);

            const int kkidx = kk >> 5;
            const int kbase = kk + sel * 16;
            #pragma unroll
            for (int nn = 0; nn < 4; ++nn) {
                v8u bu;
                if constexpr (USE_WS) {
                    const int f = ((64 + o) * 4 + nn) * 2 + kkidx;  // mat = 1 (L)
                    bu = wsv[f * 32 + lane];
                } else {
                    const int n = nn * 16 + mrow;
                    #pragma unroll
                    for (int v = 0; v < 8; ++v) {
                        float f0 = Lb[(kbase + 2 * v    ) * BS + n];
                        float f1 = Lb[(kbase + 2 * v + 1) * BS + n];
                        bu[v] = pk_bf16(f0, f1);
                    }
                }
                v16bf b = __builtin_bit_cast(v16bf, bu);
                acc[nn] = __builtin_amdgcn_wmma_f32_16x16x32_bf16(
                    false, a, false, b, (short)0, acc[nn], false, false);
            }
        }

        // ---- epilogue: + bias, coalesced fp32 stores
        #pragma unroll
        for (int nn = 0; nn < 4; ++nn) {
            const int col = o * BS + nn * 16 + mrow;
            const float bv = bias[col];
            #pragma unroll
            for (int r = 0; r < 8; ++r) {
                const int m = r + sel * 8;
                out[(tok0 + m) * (long)IN_DIM + col] = acc[nn][r] + bv;
            }
        }
    }
}

extern "C" void kernel_launch(void* const* d_in, const int* in_sizes, int n_in,
                              void* d_out, int out_size, void* d_ws, size_t ws_size,
                              hipStream_t stream) {
    const float* x    = (const float*)d_in[0];
    const float* R    = (const float*)d_in[1];
    const float* L    = (const float*)d_in[2];
    const float* bias = (const float*)d_in[3];
    float* out = (float*)d_out;

    const long n_tok = (long)in_sizes[0] / IN_DIM;   // 32768 for the reference shapes
    const int  grid  = (int)(n_tok / M_TILE);        // 2048 workgroups
    const size_t lds = LDS_BYTES;

    const bool use_ws = (d_ws != nullptr) && (ws_size >= (size_t)WS_BYTES);
    if (use_ws) {
        preconv_b_kernel<<<128, 256, 0, stream>>>(R, L, (unsigned*)d_ws);
        monarch_fused_kernel<true><<<grid, 256, lds, stream>>>(
            x, R, L, bias, (const unsigned*)d_ws, out);
    } else {
        monarch_fused_kernel<false><<<grid, 256, lds, stream>>>(
            x, R, L, bias, nullptr, out);
    }
}
